// UncertaintyPropagationGraph_3384434230093
// MI455X (gfx1250) — matrix operand order repaired and checked by reading, not compiled, not run on hardware
//
#include <hip/hip_runtime.h>
#include <hip/hip_bf16.h>

// ---------------------------------------------------------------------------
// UncertaintyPropagationGraph on MI455X (gfx1250)
//   - all GEMMs via v_wmma_f32_16x16x32_f16, 16x64 C-tile per wave (4 acc)
//   - weights pre-converted once to f16, pre-transposed to (P x K) so LDS
//     weight tiles are contiguous and staged with global_load_async_to_lds
// Shapes: B=512, M=16, D=256, L=3, T=4, H=256, N=B*M=8192
// ---------------------------------------------------------------------------

typedef __attribute__((ext_vector_type(16))) _Float16 v16h;
typedef __attribute__((ext_vector_type(8)))  float    v8f;

#define NB      512
#define NM      16
#define ND      256
#define NL      3
#define NT      4
#define NROWS   8192            // B*M
#define HISTLD  1024            // T*D   (row length inside history buffer)
#define HISTSB  16384           // M*T*D (per-b stride in history buffer)

__device__ __forceinline__ float sigmoidf_(float x) {
    return 1.0f / (1.0f + __expf(-x));
}

// ---------------------------------------------------------------------------
// WMMA GEMM:  out[r, col] (+)= sum_k A[r,k] * Wt[col, k]
//   A row address   = (r/16)*aStrideB + (r%16)*aLd + aBase     (f32)
//   out row address = (r/16)*oStrideB + (r%16)*oLd + oBase     (f32)
//   Wt: f16, (P x K) row-major (pre-transposed weight), so a 64-column LDS
//       tile is one contiguous block -> staged via async copy to LDS.
// Block: 256 threads = 8 waves. Block tile 128 rows x 64 cols.
// Wave tile 16 x 64: one A-fragment feeds 4 WMMAs per 32-deep k-chunk.
// grid = (N/128, P/64).  LDS tile = 64*K f16 (K=256 -> 32KB, K=512 -> 64KB).
// ---------------------------------------------------------------------------
__global__ void wmma_gemm_kernel(const float* __restrict__ A,
                                 int aStrideB, int aLd, int aBase,
                                 const _Float16* __restrict__ Wt,
                                 const float* __restrict__ bias1,
                                 const float* __restrict__ bias2,
                                 float* __restrict__ out,
                                 int oStrideB, int oLd, int oBase,
                                 int K, int relu, int accum) {
    extern __shared__ __align__(32) char smemRaw[];
    _Float16* smem = (_Float16*)smemRaw;

    const int tid  = threadIdx.x;
    const int col0 = blockIdx.y * 64;

    // ---- stage 64 weight rows of Wt (= 64 output columns) into LDS -------
    // contiguous global block: Wt + col0*K, 64*K*2 bytes, b128 per lane/iter
    {
        const char* gsrc = (const char*)(Wt + (size_t)col0 * K);
        unsigned ldsBase = (unsigned)(unsigned long long)(void*)smemRaw;
        const int iters = K >> 5;                 // (64*K*2) / (256*16)
        for (int it = 0; it < iters; ++it) {
            unsigned loff = ldsBase + (unsigned)(tid * 16 + it * 4096);
            unsigned long long ga =
                (unsigned long long)(gsrc + tid * 16 + it * 4096);
            asm volatile("global_load_async_to_lds_b128 %0, %1, off"
                         :: "v"(loff), "v"(ga) : "memory");
        }
        asm volatile("s_wait_asynccnt 0x0" ::: "memory");
    }
    __syncthreads();

    const int wid  = tid >> 5;
    const int lane = tid & 31;
    const int r0   = (blockIdx.x * 8 + wid) * 16;
    const int hi16 = (lane & 16) ? 1 : 0;

    // A-fragment K-offsets (16-bit A 16x32 layout):
    //   lanes 0-15: K = c*32 + {0..7, 16..23};  lanes 16-31: +8
    const int ksel  = hi16 ? 8 : 0;
    // B-fragment K-offsets (same as A): lanes 0-15: K = c*32 + {0..7, 16..23}; lanes 16-31: +8
    const int kksel = hi16 ? 8 : 0;
    const int jj    = lane & 15;

    const int row = r0 + jj;
    const size_t aoff =
        (size_t)(row >> 4) * aStrideB + (size_t)(row & 15) * aLd + aBase;

    v8f acc[4] = {};
    const int nc = K >> 5;
    for (int c = 0; c < nc; ++c) {
        const float* ap = A + aoff + c * 32 + ksel;
        float4 a0 = *(const float4*)(ap);
        float4 a1 = *(const float4*)(ap + 4);
        float4 a2 = *(const float4*)(ap + 16);
        float4 a3 = *(const float4*)(ap + 20);
        if (c + 1 < nc) __builtin_prefetch(ap + 32, 0, 3);

        float av[16] = {a0.x, a0.y, a0.z, a0.w, a1.x, a1.y, a1.z, a1.w,
                        a2.x, a2.y, a2.z, a2.w, a3.x, a3.y, a3.z, a3.w};
        v16h af;
#pragma unroll
        for (int i = 0; i < 16; ++i) af[i] = (_Float16)av[i];

#pragma unroll
        for (int s = 0; s < 4; ++s) {
            const _Float16* bp =
                smem + (size_t)(s * 16 + jj) * K + c * 32 + kksel;
            v16h bf;
#pragma unroll
            for (int i = 0; i < 16; ++i) bf[i] = bp[(i < 8) ? i : i + 8];
            acc[s] = __builtin_amdgcn_wmma_f32_16x16x32_f16(
                false, af, false, bf, (short)0, acc[s], false, false);
        }
    }

    // ---- epilogue: C tile: VGPR v holds row r0+v (+8 for hi lanes) -------
    const int rsel = hi16 ? 8 : 0;
#pragma unroll
    for (int s = 0; s < 4; ++s) {
        const int col = col0 + s * 16 + jj;
        float bv = 0.0f;
        if (bias1) bv += bias1[col];
        if (bias2) bv += bias2[col];
#pragma unroll
        for (int v = 0; v < 8; ++v) {
            int r = r0 + rsel + v;
            size_t o = (size_t)(r >> 4) * oStrideB +
                       (size_t)(r & 15) * oLd + oBase + col;
            float val = acc[s][v] + bv;
            if (relu) val = fmaxf(val, 0.0f);
            if (accum) val += out[o];
            out[o] = val;
        }
    }
}

// ---------------------------------------------------------------------------
// Weight prep: f32 (K x P) -> f16 (P x K)  (transpose + convert, run once)
// ---------------------------------------------------------------------------
__global__ void wprep_t_kernel(const float* __restrict__ in,
                               _Float16* __restrict__ out, int K, int P) {
    int idx = blockIdx.x * 256 + threadIdx.x;      // 0 .. K*P-1
    int p = idx / K;
    int k = idx - p * K;
    out[idx] = (_Float16)in[(size_t)k * P + p];
}

// elementwise f32 -> f16 (layout already (P x K))
__global__ void wprep_c_kernel(const float* __restrict__ in,
                               _Float16* __restrict__ out) {
    int idx = blockIdx.x * 256 + threadIdx.x;
    out[idx] = (_Float16)in[idx];
}

// ---------------------------------------------------------------------------
// x0 = mode_features + mode_emb  -> history slot 0
// ---------------------------------------------------------------------------
__global__ void embed_kernel(const float* __restrict__ mf,
                             const float* __restrict__ memb,
                             float* __restrict__ hist) {
    int idx = blockIdx.x * 256 + threadIdx.x;      // 0 .. B*M*D-1
    int b   = idx >> 12;                            // /(M*D)
    int md  = idx & 4095;
    hist[(size_t)b * HISTSB + md] = mf[idx] + memb[md];
}

// ---------------------------------------------------------------------------
// Edge weights: wmask[b,i,j] = sigmoid(relu(hi[b,i]+hj[b,j]+eb1) . eW2 + eb2)
//               * (adj[i,j] != 0)
// One wave per (b,i,j); 256-d reduction: 8 elems/lane + xor-shuffle tree.
// ---------------------------------------------------------------------------
__global__ void edge_kernel(const float* __restrict__ hi,
                            const float* __restrict__ hj,
                            const float* __restrict__ eb1,
                            const float* __restrict__ eW2,
                            const float* __restrict__ eb2,
                            const int*   __restrict__ adj,
                            float* __restrict__ wmask) {
    int g    = blockIdx.x * 8 + (threadIdx.x >> 5);   // (b,i,j) flat
    int lane = threadIdx.x & 31;
    int b    = g >> 8;
    int rem  = g & 255;
    int i    = rem >> 4;
    int j    = rem & 15;

    const float* hp = hi + (size_t)(b * 16 + i) * ND + lane * 8;
    const float* gp = hj + (size_t)(b * 16 + j) * ND + lane * 8;
    const float* bp = eb1 + lane * 8;
    const float* wp = eW2 + lane * 8;

    float4 h0 = *(const float4*)(hp),     h1 = *(const float4*)(hp + 4);
    float4 g0 = *(const float4*)(gp),     g1 = *(const float4*)(gp + 4);
    float4 b0 = *(const float4*)(bp),     b1 = *(const float4*)(bp + 4);
    float4 w0 = *(const float4*)(wp),     w1 = *(const float4*)(wp + 4);

    float hv[8] = {h0.x, h0.y, h0.z, h0.w, h1.x, h1.y, h1.z, h1.w};
    float gv[8] = {g0.x, g0.y, g0.z, g0.w, g1.x, g1.y, g1.z, g1.w};
    float bv[8] = {b0.x, b0.y, b0.z, b0.w, b1.x, b1.y, b1.z, b1.w};
    float wv[8] = {w0.x, w0.y, w0.z, w0.w, w1.x, w1.y, w1.z, w1.w};

    float p = 0.0f;
#pragma unroll
    for (int t = 0; t < 8; ++t) {
        float r = fmaxf(hv[t] + gv[t] + bv[t], 0.0f);
        p += r * wv[t];
    }
#pragma unroll
    for (int off = 16; off > 0; off >>= 1) p += __shfl_xor(p, off, 32);

    if (lane == 0) {
        float w = sigmoidf_(p + eb2[0]);
        if (adj[i * 16 + j] == 0) w = 0.0f;
        wmask[g] = w;
    }
}

// ---------------------------------------------------------------------------
// agg[b,i] = sum_j wmask[b,i,j] * x[b,j]   (fallback: x[b,i] if row i has no
// neighbors), then ni = concat(x, agg) -> (N x 512).  One block per b.
// ---------------------------------------------------------------------------
__global__ void agg_concat_kernel(const float* __restrict__ hist, int histBase,
                                  const float* __restrict__ wmask,
                                  const int*   __restrict__ adj,
                                  float* __restrict__ ni) {
    __shared__ float xs[16 * ND];
    __shared__ float wsd[256];
    __shared__ int   hasnb[16];

    int b   = blockIdx.x;
    int tid = threadIdx.x;

    for (int idx = tid; idx < 16 * ND; idx += 256)
        xs[idx] = hist[(size_t)b * HISTSB + histBase + idx];
    wsd[tid] = wmask[(size_t)b * 256 + tid];
    if (tid < 16) {
        int any = 0;
#pragma unroll
        for (int j = 0; j < 16; ++j) any |= (adj[tid * 16 + j] != 0);
        hasnb[tid] = any;
    }
    __syncthreads();

    int i  = tid >> 4;
    int d0 = (tid & 15) * 16;
    float s[16];
#pragma unroll
    for (int dd = 0; dd < 16; ++dd) s[dd] = 0.0f;
    for (int j = 0; j < 16; ++j) {
        float w = wsd[i * 16 + j];
#pragma unroll
        for (int dd = 0; dd < 16; ++dd) s[dd] += w * xs[j * ND + d0 + dd];
    }
    size_t r = (size_t)b * 16 + i;
#pragma unroll
    for (int dd = 0; dd < 16; ++dd) {
        float xv = xs[i * ND + d0 + dd];
        float av = hasnb[i] ? s[dd] : xv;
        ni[r * 512 + d0 + dd]       = xv;
        ni[r * 512 + 256 + d0 + dd] = av;
    }
}

// ---------------------------------------------------------------------------
// LSTM pointwise gates: gbuf (N x 1024) -> h,c; h also streamed to the
// sequence-layout output buffer at [n*1024 + t*256 + d].
// ---------------------------------------------------------------------------
__global__ void lstm_point_kernel(const float* __restrict__ gbuf,
                                  float* __restrict__ hstate,
                                  float* __restrict__ cstate,
                                  float* __restrict__ outseq, int tstep) {
    int idx = blockIdx.x * 256 + threadIdx.x;       // 0 .. N*D-1
    int n   = idx >> 8;
    int d   = idx & 255;
    size_t gb = (size_t)n * 1024;
    float gi = gbuf[gb + d];
    float gf = gbuf[gb + 256 + d];
    float gg = gbuf[gb + 512 + d];
    float go = gbuf[gb + 768 + d];
    float c  = sigmoidf_(gf) * cstate[idx] + sigmoidf_(gi) * tanhf(gg);
    float h  = sigmoidf_(go) * tanhf(c);
    cstate[idx] = c;
    hstate[idx] = h;
    outseq[(size_t)n * HISTLD + tstep * ND + d] = h;
}

__global__ void zero_kernel(float* __restrict__ p) {
    p[blockIdx.x * 256 + threadIdx.x] = 0.0f;
}

// final x: d_out[r*256+d] = hist slot L
__global__ void copyx_kernel(const float* __restrict__ hist,
                             float* __restrict__ out) {
    int idx = blockIdx.x * 256 + threadIdx.x;       // 0 .. B*M*D-1
    out[idx] = hist[(size_t)(idx >> 12) * HISTSB + NL * 4096 + (idx & 4095)];
}

// ---------------------------------------------------------------------------
extern "C" void kernel_launch(void* const* d_in, const int* in_sizes, int n_in,
                              void* d_out, int out_size, void* d_ws, size_t ws_size,
                              hipStream_t stream) {
    (void)in_sizes; (void)n_in; (void)out_size; (void)ws_size;

    const float* mf   = (const float*)d_in[0];
    const int*   adj  = (const int*)  d_in[1];
    const float* memb = (const float*)d_in[2];
    const float* gW1  = (const float*)d_in[3];
    const float* gb1  = (const float*)d_in[4];
    const float* gW2  = (const float*)d_in[5];
    const float* gb2  = (const float*)d_in[6];
    const float* eW1  = (const float*)d_in[7];
    const float* eb1  = (const float*)d_in[8];
    const float* eW2  = (const float*)d_in[9];
    const float* eb2  = (const float*)d_in[10];
    const float* lWih = (const float*)d_in[11];
    const float* lWhh = (const float*)d_in[12];
    const float* lbih = (const float*)d_in[13];
    const float* lbhh = (const float*)d_in[14];

    float* ws = (float*)d_ws;
    // ---- f32 workspace layout (floats) ----
    float* hist   = ws;                         // 8,388,608  (B,T,M,D) flat
    float* hi     = ws + 8388608;               // 2,097,152
    float* hj     = hi + 2097152;               // 2,097,152
    float* wmask  = hj + 2097152;               //   131,072
    float* ni     = wmask + 131072;             // 4,194,304
    float* t1     = ni + 4194304;               // 2,097,152
    float* gbuf   = hi;                         // 8,388,608 (aliases hi..ni)
    float* l0out  = t1 + 2097152;               // 8,388,608
    float* hstate = l0out + 8388608;            // 2,097,152
    float* cstate = hstate + 2097152;           // 2,097,152
    // ---- f16 pre-transposed weights (halves) ----
    _Float16* wh     = (_Float16*)(cstate + 2097152);
    _Float16* eW1aT  = wh;                      //  65,536  (256x256)
    _Float16* eW1bT  = eW1aT + 65536;           //  65,536
    _Float16* gW1T   = eW1bT + 65536;           // 3x131,072 (256x512)
    _Float16* gW2T   = gW1T + 3 * 131072;       // 3x 65,536 (256x256)
    _Float16* lWihH  = gW2T + 3 * 65536;        // 2x262,144 (1024x256)
    _Float16* lWhhH  = lWihH + 2 * 262144;      // 2x262,144

    float* out_x   = (float*)d_out;             // (B,M,D)
    float* out_dyn = out_x + 2097152;           // (B,M,T,D) == ys flat

    auto gemm = [&](const float* A, int asb, int ald, int abase,
                    const _Float16* Wt, const float* b1, const float* b2,
                    float* out, int osb, int old_, int obase,
                    int K, int P, int relu, int accum) {
        dim3 grid(NROWS / 128, P / 64);
        dim3 blk(256);
        size_t sh = (size_t)K * 64 * sizeof(_Float16);   // 32KB / 64KB
        wmma_gemm_kernel<<<grid, blk, sh, stream>>>(
            A, asb, ald, abase, Wt, b1, b2, out, osb, old_, obase,
            K, relu, accum);
    };

    // 0) one-time weight prep: f16, (P x K) layout
    wprep_t_kernel<<<256, 256, 0, stream>>>(eW1,            eW1aT, ND, ND);
    wprep_t_kernel<<<256, 256, 0, stream>>>(eW1 + ND * ND,  eW1bT, ND, ND);
    for (int l = 0; l < NL; ++l) {
        wprep_t_kernel<<<512, 256, 0, stream>>>(gW1 + (size_t)l * 512 * ND,
                                                gW1T + (size_t)l * 131072, 512, ND);
        wprep_t_kernel<<<256, 256, 0, stream>>>(gW2 + (size_t)l * ND * ND,
                                                gW2T + (size_t)l * 65536, ND, ND);
    }
    wprep_c_kernel<<<2048, 256, 0, stream>>>(lWih, lWihH);   // already (P x K)
    wprep_c_kernel<<<2048, 256, 0, stream>>>(lWhh, lWhhH);

    // 1) x0 = mode_features + mode_emb -> hist slot 0
    embed_kernel<<<NROWS, 256, 0, stream>>>(mf, memb, hist);

    // 2) graph layers
    for (int l = 0; l < NL; ++l) {
        const int xbase = l * 4096;  // hist slot l
        gemm(hist, HISTSB, ND, xbase, eW1aT, nullptr, nullptr,
             hi, 16 * ND, ND, 0, ND, ND, 0, 0);
        gemm(hist, HISTSB, ND, xbase, eW1bT, nullptr, nullptr,
             hj, 16 * ND, ND, 0, ND, ND, 0, 0);
        edge_kernel<<<(NB * NM * NM) / 8, 256, 0, stream>>>(
            hi, hj, eb1, eW2, eb2, adj, wmask);
        agg_concat_kernel<<<NB, 256, 0, stream>>>(hist, xbase, wmask, adj, ni);
        // t1 = relu(ni @ gW1[l] + gb1[l])
        gemm(ni, 16 * 512, 512, 0, gW1T + (size_t)l * 131072,
             gb1 + l * ND, nullptr, t1, 16 * ND, ND, 0, 512, ND, 1, 0);
        // x_{l+1} = t1 @ gW2[l] + gb2[l]  -> hist slot l+1
        gemm(t1, 16 * ND, ND, 0, gW2T + (size_t)l * 65536,
             gb2 + l * ND, nullptr, hist, HISTSB, ND, (l + 1) * 4096,
             ND, ND, 0, 0);
    }

    // 3) final x -> d_out
    copyx_kernel<<<NROWS, 256, 0, stream>>>(hist, out_x);

    // 4) two LSTM layers over 8192 sequences of length T=4
    for (int k = 0; k < 2; ++k) {
        const _Float16* WihH = lWihH + (size_t)k * 262144;  // (1024x256)
        const _Float16* WhhH = lWhhH + (size_t)k * 262144;
        const float* bih = lbih + k * 1024;
        const float* bhh = lbhh + k * 1024;
        const float* xseq = (k == 0) ? hist : l0out;        // [n*1024+t*256+d]
        float* outseq = (k == 0) ? l0out : out_dyn;

        zero_kernel<<<(2 * 2097152) / 256, 256, 0, stream>>>(hstate); // h & c
        for (int t = 0; t < NT; ++t) {
            // gbuf = x_t @ Wih^T + bih + bhh
            gemm(xseq, HISTSB, HISTLD, t * ND, WihH, bih, bhh,
                 gbuf, 16 * 1024, 1024, 0, ND, 1024, 0, 0);
            // gbuf += h @ Whh^T
            gemm(hstate, 16 * ND, ND, 0, WhhH, nullptr, nullptr,
                 gbuf, 16 * 1024, 1024, 0, ND, 1024, 0, 1);
            // gates -> h, c, outseq[:, t, :]
            lstm_point_kernel<<<NROWS, 256, 0, stream>>>(
                gbuf, hstate, cstate, outseq, t);
        }
    }
}
